// dyMEAN_45655502357202
// MI455X (gfx1250) — matrix operand order repaired and checked by reading, not compiled, not run on hardware
//
#include <hip/hip_runtime.h>
#include <hip/hip_bf16.h>

// ---------------- types & helpers ----------------
typedef __attribute__((ext_vector_type(16))) _Float16 v16h;
typedef __attribute__((ext_vector_type(8)))  _Float16 v8h;
typedef __attribute__((ext_vector_type(8)))  float    v8f;
typedef __attribute__((ext_vector_type(4)))  int      v4i;

#define DEV static __device__ __forceinline__

#if defined(__AMDGCN__) && __has_builtin(__builtin_amdgcn_global_load_async_to_lds_b128) && __has_builtin(__builtin_amdgcn_s_wait_asynccnt)
#define HAVE_ASYNC 1
typedef __attribute__((address_space(1))) v4i gv4i;   // global (AS1) int4
typedef __attribute__((address_space(3))) v4i lv4i;   // LDS (AS3) int4
#else
#define HAVE_ASYNC 0
#endif

DEV float fast_rcp(float x) {
#if defined(__AMDGCN__) && __has_builtin(__builtin_amdgcn_rcpf)
    return __builtin_amdgcn_rcpf(x);
#else
    return 1.0f / x;
#endif
}
DEV float silu_f(float x) { return x * fast_rcp(1.0f + __expf(-x)); }

DEV v8f wmma_f16(v16h a, v16h b, v8f c) {
    // D = A(16x32 f16) * B(32x16 f16) + C(16x16 f32)
    return __builtin_amdgcn_wmma_f32_16x16x32_f16(
        false, a, false, b, (short)0, c, false, false);
}

// Build an A fragment from a row of f16 data. `p` must already point at
// row_base + kb + ((lane>=16)?8:0).  Halves i<8 -> k=klo+i ; i>=8 -> k=klo+16+(i-8).
DEV v16h frag_from_row(const _Float16* p) {
    v8h lo = *(const v8h*)(p);
    v8h hi = *(const v8h*)(p + 16);
    v16h r;
#pragma unroll
    for (int i = 0; i < 8; ++i) { r[i] = lo[i]; r[i + 8] = hi[i]; }
    return r;
}

// B fragment from pre-swizzled weights: tile = (kt*ntiles+nt), 32 lanes x 16 halfs.
DEV v16h load_bfrag(const _Float16* w, int ntiles, int kt, int nt, int lane) {
    return *(const v16h*)(w + (((size_t)(kt * ntiles + nt)) << 9) + (lane << 4));
}

// 16-byte async global->LDS copy (gfx1250 GLOBAL_LOAD_ASYNC_TO_LDS_B128)
DEV void async_cp16(const void* g, void* l) {
#if HAVE_ASYNC
    __builtin_amdgcn_global_load_async_to_lds_b128((gv4i*)g, (lv4i*)l, 0, 0);
#endif
}
DEV void async_wait0() {
#if HAVE_ASYNC
    __builtin_amdgcn_s_wait_asynccnt(0);
#endif
}

// simple hash RNG (stand-in for JAX PRNG; no runtime validation available)
DEV unsigned pcg(unsigned x) {
    x = x * 747796405u + 2891336453u;
    unsigned w = ((x >> ((x >> 28) + 4u)) ^ x) * 277803737u;
    return (w >> 22) ^ w;
}
DEV float rnd_norm(unsigned seed, unsigned idx) {
    unsigned a = pcg(seed ^ (idx * 2654435761u));
    unsigned b = pcg(a + 0x9E3779B9u);
    float u1 = (a + 1u) * 2.3283064e-10f;
    float u2 = b * 2.3283064e-10f;
    return sqrtf(-2.0f * __logf(u1 + 1e-12f)) * __cosf(6.2831853f * u2);
}

#define NNODES 768
#define NEGV  -1000000000.0f

// ---------------- weight swizzle into WMMA B-fragment order ----------------
__global__ void k_swz(const float* __restrict__ W, _Float16* __restrict__ out,
                      int K, int N) {
    int idx = blockIdx.x * blockDim.x + threadIdx.x;
    if (idx >= K * N) return;
    int ntiles = N >> 4;
    int tile = idx >> 9, r = idx & 511;
    int lane = r >> 4, i = r & 15;
    int kt = tile / ntiles, nt = tile - kt * ntiles;
    int k = kt * 32 + ((lane >= 16) ? 8 : 0) + ((i < 8) ? i : (16 + i - 8));
    int n = nt * 16 + (lane & 15);
    out[idx] = (_Float16)W[(size_t)k * N + n];
}

// ---------------- setup kernels ----------------
__global__ void k_centroid(const float* __restrict__ X,
                           const unsigned char* __restrict__ mask,
                           float* __restrict__ cen) {
    __shared__ float s[4];
    int b = blockIdx.x, t = threadIdx.x;              // 96 threads
    if (t < 4) s[t] = 0.f;
    __syncthreads();
    int nn = b * 96 + t;
    float wv = mask[nn] ? 0.f : 1.f;
#pragma unroll
    for (int c = 0; c < 3; ++c)
        atomicAdd(&s[c], X[(size_t)nn * 42 + 3 + c] * wv);  // CA channel == 1
    atomicAdd(&s[3], wv);
    __syncthreads();
    if (t < 3) cen[b * 3 + t] = s[t] / fmaxf(s[3], 1.f);
}

__global__ void k_initmask(const float* __restrict__ X, const int* __restrict__ S,
                           const unsigned char* __restrict__ mask,
                           const unsigned char* __restrict__ amask,
                           const float* __restrict__ cen,
                           float* __restrict__ Xa, int* __restrict__ Sw,
                           float* __restrict__ aw) {
    int nn = blockIdx.x, t = threadIdx.x;             // 64 threads
    int b = nn / 96;
    unsigned char mk = mask[nn];
    if (t == 0) Sw[nn] = mk ? 21 : S[nn];
    if (t < 14) aw[nn * 14 + t] = (amask[nn * 14 + t] || mk) ? 1.f : 0.f;
    if (t < 42) {
        float v;
        if (mk) {
            int c = t % 3;
            float ca = rnd_norm(0x1234u, nn * 3 + c) * 10.f + cen[b * 3 + c];
            v = ca + rnd_norm(0x5678u, nn * 42 + t);
        } else v = X[(size_t)nn * 42 + t];
        Xa[(size_t)nn * 42 + t] = v;
    }
}

__global__ void k_normstats(const float* __restrict__ Xa,
                            const unsigned char* __restrict__ mask,
                            float* __restrict__ centers) {
    __shared__ float s[43];
    int b = blockIdx.x, t = threadIdx.x;              // 96 threads
    if (t < 43) s[t] = 0.f;
    __syncthreads();
    int nn = b * 96 + t;
    float wv = mask[nn] ? 0.f : 1.f;
    for (int k = 0; k < 42; ++k)
        atomicAdd(&s[k], Xa[(size_t)nn * 42 + k] * wv);
    atomicAdd(&s[42], wv);
    __syncthreads();
    if (t < 3) {
        float den = fmaxf(s[42], 1.f), a = 0.f;
        for (int ch = 0; ch < 14; ++ch) a += s[ch * 3 + t];
        centers[b * 3 + t] = a / (14.f * den);
    }
}

__global__ void k_normalize(float* __restrict__ Xa,
                            const float* __restrict__ centers) {
    int nn = blockIdx.x, t = threadIdx.x;             // 64 threads
    if (t < 42) {
        int b = nn / 96;
        size_t o = (size_t)nn * 42 + t;
        Xa[o] = (Xa[o] - centers[b * 3 + t % 3]) * 0.1f;
    }
}

// channel attention softmax (per node, all 3 layers)
__global__ void k_attn(const int* __restrict__ Sw,
                       const float* __restrict__ atom_embed,
                       const float* __restrict__ attnW,
                       const float* __restrict__ aw,
                       float* __restrict__ attn_out) {
    __shared__ float lg[56];
    int nn = blockIdx.x, t = threadIdx.x;             // 64 threads
    int s = Sw[nn];
    for (int l = 0; l < 3; ++l) {
        if (t < 56) {
            int ch = t >> 2, k = t & 3;
            const float* ce = atom_embed + ((size_t)s * 14 + ch) * 32;
            float acc = 0.f;
            for (int j = 0; j < 32; ++j) acc += ce[j] * attnW[(l * 32 + j) * 4 + k];
            lg[t] = acc;
        }
        __syncthreads();
        if (t < 4) {
            float mx = -1e30f;
            for (int ch = 0; ch < 14; ++ch) {
                float v = (aw[nn * 14 + ch] > 0.f) ? lg[ch * 4 + t] : NEGV;
                mx = fmaxf(mx, v);
            }
            float sum = 0.f;
            for (int ch = 0; ch < 14; ++ch) {
                float v = (aw[nn * 14 + ch] > 0.f) ? lg[ch * 4 + t] : NEGV;
                sum += __expf(v - mx);
            }
            float inv = fast_rcp(sum);
            for (int ch = 0; ch < 14; ++ch) {
                float v = (aw[nn * 14 + ch] > 0.f) ? lg[ch * 4 + t] : NEGV;
                attn_out[((size_t)l * NNODES + nn) * 56 + ch * 4 + t] =
                    __expf(v - mx) * inv;
            }
        }
        __syncthreads();
    }
}

// H0 = (smooth@res_embed | res_embed[S]) + pos_embed
__global__ void k_embed(const int* __restrict__ Sw,
                        const unsigned char* __restrict__ mask,
                        const int* __restrict__ pid,
                        const float* __restrict__ res_embed,
                        const float* __restrict__ pos_embed,
                        const float* __restrict__ smooth, int use_smooth,
                        float* __restrict__ H0, _Float16* __restrict__ H0h) {
    int nn = blockIdx.x, c = threadIdx.x;             // 128 threads
    float v;
    if (use_smooth && mask[nn]) {
        v = 0.f;
        for (int k = 0; k < 25; ++k)
            v += smooth[nn * 25 + k] * res_embed[(size_t)k * 128 + c];
    } else {
        v = res_embed[(size_t)Sw[nn] * 128 + c];
    }
    v += pos_embed[(size_t)pid[nn] * 128 + c];
    size_t o = (size_t)nn * 128 + c;
    H0[o] = v;
    H0h[o] = (_Float16)v;
}

// ---------------- generic 768x128 @ 128x128 WMMA GEMM ----------------
__global__ __launch_bounds__(256)
void k_lin128(const _Float16* __restrict__ A16, const _Float16* __restrict__ ws,
              const float* __restrict__ bias, const float* __restrict__ addTo,
              float* __restrict__ outF, _Float16* __restrict__ outH,
              int act_silu, int outh_silu) {
    __shared__ _Float16 shA[16 * 128];
    int tid = threadIdx.x, lane = tid & 31, w = tid >> 5;
    int n0 = blockIdx.x * 16;
#if HAVE_ASYNC
    // 4KB tile: one async b128 per thread, tracked by ASYNCcnt
    async_cp16((const char*)(A16 + (size_t)n0 * 128) + tid * 16,
               (char*)shA + tid * 16);
    async_wait0();
#else
    for (int idx = tid; idx < 2048; idx += 256)
        shA[idx] = A16[(size_t)n0 * 128 + idx];
#endif
    __builtin_prefetch(ws + (tid << 5), 0, 3);
    __syncthreads();
    int n = (w << 4) + (lane & 15);
    int sub = ((lane >> 4) << 3);
    v8f acc = {0.f, 0.f, 0.f, 0.f, 0.f, 0.f, 0.f, 0.f};
#pragma unroll
    for (int kb = 0; kb < 4; ++kb) {
        v16h af = frag_from_row(shA + (lane & 15) * 128 + kb * 32 + sub);
        v16h bf = load_bfrag(ws, 8, kb, w, lane);
        acc = wmma_f16(af, bf, acc);
    }
    float bv = bias[n];
    int rb = (lane >> 4) << 3;
#pragma unroll
    for (int v = 0; v < 8; ++v) {
        float val = acc[v] + bv;
        if (act_silu) val = silu_f(val);
        size_t o = (size_t)(n0 + rb + v) * 128 + n;
        if (addTo) val += addTo[o];
        if (outF) outF[o] = val;
        if (outH) outH[o] = (_Float16)(outh_silu ? silu_f(val) : val);
    }
}

// ---------------- per-node edge layer (96 edges, fused GEMM chain) ----------------
__global__ __launch_bounds__(256)
void k_edge_layer(const float* __restrict__ Xin, float* __restrict__ Xout,
                  const _Float16* __restrict__ g_h16,
                  const float* __restrict__ attnL,     // layer slice [N][56]
                  const float* __restrict__ aw,
                  const float* __restrict__ radW, const float* __restrict__ radB,
                  const _Float16* __restrict__ w1s, const float* __restrict__ b1,
                  const _Float16* __restrict__ w2s, const float* __restrict__ b2,
                  const _Float16* __restrict__ cw1s, const float* __restrict__ cb1,
                  const float* __restrict__ cw2,
                  float* __restrict__ g_agg) {
    __shared__ _Float16 bufA[96 * 128];   // rad (f16) -> later m (f16)
    __shared__ _Float16 bufB[96 * 128];   // pooled (f32 alias) -> later m1 (f16)
    __shared__ _Float16 sh_d[96 * 42];    // edge deltas (f16)
    __shared__ float    sh_cw[96];
    float* sh_pool = (float*)bufB;        // 96*16 f32 fits in bufB, dead before m1

    const int tid = threadIdx.x;
    const int lane = tid & 31;
    const int w = tid >> 5;
    const int i = blockIdx.x;             // node
    const int b = i / 96;
    const int bb = b * 96;

    // prefetch hot weight streams (L2-resident, reused by all 768 blocks)
    __builtin_prefetch(w1s + (tid << 5), 0, 3);
    __builtin_prefetch(w2s + (tid << 4), 0, 3);
    __builtin_prefetch(cw1s + (tid << 4), 0, 3);

    // ---- phase 1: geometry (d, pooled) ----
    if (tid < 96) {
        const int e = tid;
        const float* Xi = Xin + (size_t)i * 42;
        const float* Xj = Xin + (size_t)(bb + e) * 42;
        float d[42];
#pragma unroll
        for (int t = 0; t < 42; ++t) {
            d[t] = Xi[t] - Xj[t];
            sh_d[e * 42 + t] = (_Float16)d[t];
        }
        float G[9];
#pragma unroll
        for (int t = 0; t < 9; ++t) G[t] = 0.f;
        for (int ch = 0; ch < 14; ++ch)
#pragma unroll
            for (int a = 0; a < 3; ++a)
#pragma unroll
                for (int c = 0; c < 3; ++c)
                    G[a * 3 + c] += d[ch * 3 + a] * d[ch * 3 + c];
        float fro2 = 0.f;
#pragma unroll
        for (int t = 0; t < 9; ++t) fro2 += G[t] * G[t];
        float inv = fast_rcp(sqrtf(fro2) + 1.f);
        const float* ar = attnL + (size_t)i * 56;
        const float* ac = attnL + (size_t)(bb + e) * 56;
        float t1[12], t2[12];
#pragma unroll
        for (int t = 0; t < 12; ++t) { t1[t] = 0.f; t2[t] = 0.f; }
        for (int ch = 0; ch < 14; ++ch)
#pragma unroll
            for (int k = 0; k < 4; ++k) {
                float a0 = ar[ch * 4 + k], a1 = ac[ch * 4 + k];
#pragma unroll
                for (int c = 0; c < 3; ++c) {
                    t1[k * 3 + c] += a0 * d[ch * 3 + c];
                    t2[k * 3 + c] += a1 * d[ch * 3 + c];
                }
            }
#pragma unroll
        for (int k = 0; k < 4; ++k)
#pragma unroll
            for (int k2 = 0; k2 < 4; ++k2) {
                float s = 0.f;
#pragma unroll
                for (int c = 0; c < 3; ++c) s += t1[k * 3 + c] * t2[k2 * 3 + c];
                sh_pool[e * 16 + k * 4 + k2] = s * inv;
            }
    }
    __syncthreads();

    // ---- phase 2: rad = silu(pooled @ rad_W + rad_b), f16 into bufA ----
    for (int idx = tid; idx < 96 * 128; idx += 256) {
        int e = idx >> 7, c = idx & 127;
        float acc = radB[c];
#pragma unroll
        for (int j = 0; j < 16; ++j) acc += sh_pool[e * 16 + j] * radW[j * 128 + c];
        bufA[idx] = (_Float16)silu_f(acc);
    }
    __syncthreads();

    const int n = (w << 4) + (lane & 15);
    const int sub = ((lane >> 4) << 3);
    const int rbo = (lane >> 4) << 3;

    // ---- phase 3: m1 = silu([h_row | h_col | rad] @ W1 + b1) ----
    {
        float bv = b1[n];
        for (int et = 0; et < 6; ++et) {
            int m0 = et * 16;
            v8f acc = {0.f, 0.f, 0.f, 0.f, 0.f, 0.f, 0.f, 0.f};
#pragma unroll
            for (int kb = 0; kb < 12; ++kb) {
                int r = kb >> 2;
                int koff = (kb - (r << 2)) * 32 + sub;
                const _Float16* ap;
                if (r == 0)
                    ap = g_h16 + (size_t)i * 128 + koff;               // h[row], same for all M
                else if (r == 1)
                    ap = g_h16 + (size_t)(bb + m0 + (lane & 15)) * 128 + koff;  // h[col]
                else
                    ap = bufA + (m0 + (lane & 15)) * 128 + koff;       // rad
                v16h af = frag_from_row(ap);
                v16h bf = load_bfrag(w1s, 8, kb, w, lane);
                acc = wmma_f16(af, bf, acc);
            }
#pragma unroll
            for (int v = 0; v < 8; ++v)
                bufB[(m0 + rbo + v) * 128 + n] = (_Float16)silu_f(acc[v] + bv);
        }
    }
    __syncthreads();

    // ---- phase 4: m = silu(m1 @ W2 + b2) -> bufA ----
    {
        float bv = b2[n];
        for (int et = 0; et < 6; ++et) {
            int m0 = et * 16;
            v8f acc = {0.f, 0.f, 0.f, 0.f, 0.f, 0.f, 0.f, 0.f};
#pragma unroll
            for (int kb = 0; kb < 4; ++kb) {
                v16h af = frag_from_row(bufB + (m0 + (lane & 15)) * 128 + kb * 32 + sub);
                v16h bf = load_bfrag(w2s, 8, kb, w, lane);
                acc = wmma_f16(af, bf, acc);
            }
#pragma unroll
            for (int v = 0; v < 8; ++v)
                bufA[(m0 + rbo + v) * 128 + n] = (_Float16)silu_f(acc[v] + bv);
        }
    }
    __syncthreads();
    if (tid < 96) sh_cw[tid] = 0.f;
    __syncthreads();

    // ---- phase 5: cw = silu(m @ coord_W1 + cb1) . coord_W2 via LDS atomics ----
    {
        float cb = cb1[n];
        float cwn = cw2[n];
        for (int et = 0; et < 6; ++et) {
            int m0 = et * 16;
            v8f acc = {0.f, 0.f, 0.f, 0.f, 0.f, 0.f, 0.f, 0.f};
#pragma unroll
            for (int kb = 0; kb < 4; ++kb) {
                v16h af = frag_from_row(bufA + (m0 + (lane & 15)) * 128 + kb * 32 + sub);
                v16h bf = load_bfrag(cw1s, 8, kb, w, lane);
                acc = wmma_f16(af, bf, acc);
            }
#pragma unroll
            for (int v = 0; v < 8; ++v)
                atomicAdd(&sh_cw[m0 + rbo + v], silu_f(acc[v] + cb) * cwn);
        }
    }
    __syncthreads();

    // ---- phase 6: node aggregation + coordinate update ----
    if (tid < 128) {
        float s = 0.f;
        for (int e = 0; e < 96; ++e) s += (float)bufA[e * 128 + tid];
        g_agg[(size_t)i * 128 + tid] = s;
    }
    if (tid >= 128 && tid < 170) {
        int t = tid - 128, ch = t / 3;
        float s = 0.f;
        for (int e = 0; e < 96; ++e) s += (float)sh_d[e * 42 + t] * sh_cw[e];
        size_t o = (size_t)i * 42 + t;
        Xout[o] = Xin[o] + (s * (1.f / 96.f)) * aw[i * 14 + ch];
    }
}

// ---------------- node update: h += MLP(cat(h, agg)) ----------------
__global__ __launch_bounds__(256)
void k_node_update(const float* __restrict__ g_agg, float* __restrict__ g_h,
                   _Float16* __restrict__ g_h16,
                   const _Float16* __restrict__ w1s, const float* __restrict__ b1v,
                   const _Float16* __restrict__ w2s, const float* __restrict__ b2v) {
    __shared__ _Float16 shin[16 * 256];
    __shared__ _Float16 shhid[16 * 128];
    int tid = threadIdx.x, lane = tid & 31, w = tid >> 5;
    int n0 = blockIdx.x * 16;
#if HAVE_ASYNC
    {   // h half of the concat tile: 16 rows x 16 chunks of 16B, async to LDS
        int r = tid >> 4, c8 = tid & 15;
        async_cp16((const char*)(g_h16 + (size_t)(n0 + r) * 128) + c8 * 16,
                   (char*)shin + r * 512 + c8 * 16);
    }
    // agg half needs f32->f16 conversion: plain loads
    for (int idx = tid; idx < 2048; idx += 256) {
        int r = idx >> 7, c = idx & 127;
        shin[r * 256 + 128 + c] = (_Float16)g_agg[(size_t)(n0 + r) * 128 + c];
    }
    async_wait0();
#else
    for (int idx = tid; idx < 2048; idx += 256) {
        int r = idx >> 7, c = idx & 127;
        shin[r * 256 + c]       = g_h16[(size_t)(n0 + r) * 128 + c];
        shin[r * 256 + 128 + c] = (_Float16)g_agg[(size_t)(n0 + r) * 128 + c];
    }
#endif
    __builtin_prefetch(w1s + (tid << 5), 0, 3);
    __syncthreads();
    int n = (w << 4) + (lane & 15);
    int sub = ((lane >> 4) << 3);
    int rb = (lane >> 4) << 3;
    v8f acc = {0.f, 0.f, 0.f, 0.f, 0.f, 0.f, 0.f, 0.f};
#pragma unroll
    for (int kb = 0; kb < 8; ++kb) {
        v16h af = frag_from_row(shin + (lane & 15) * 256 + kb * 32 + sub);
        v16h bf = load_bfrag(w1s, 8, kb, w, lane);
        acc = wmma_f16(af, bf, acc);
    }
    float bv1 = b1v[n];
#pragma unroll
    for (int v = 0; v < 8; ++v)
        shhid[(rb + v) * 128 + n] = (_Float16)silu_f(acc[v] + bv1);
    __syncthreads();
    v8f acc2 = {0.f, 0.f, 0.f, 0.f, 0.f, 0.f, 0.f, 0.f};
#pragma unroll
    for (int kb = 0; kb < 4; ++kb) {
        v16h af = frag_from_row(shhid + (lane & 15) * 128 + kb * 32 + sub);
        v16h bf = load_bfrag(w2s, 8, kb, w, lane);
        acc2 = wmma_f16(af, bf, acc2);
    }
    float bv2 = b2v[n];
#pragma unroll
    for (int v = 0; v < 8; ++v) {
        size_t o = (size_t)(n0 + rb + v) * 128 + n;
        float val = g_h[o] + acc2[v] + bv2;
        g_h[o] = val;
        g_h16[o] = (_Float16)val;
    }
}

// ---------------- logits head: res_W2 (128x25) + softmax / argmax ----------------
__global__ void k_logits(const float* __restrict__ tmpf,
                         const float* __restrict__ W2, const float* __restrict__ b2v,
                         const unsigned char* __restrict__ mask,
                         const int* __restrict__ Sw,
                         float* __restrict__ logits, float* __restrict__ smooth,
                         int* __restrict__ Sfin, int last) {
    __shared__ float sv[25];
    __shared__ float red[2];
    int nn = blockIdx.x, t = threadIdx.x;             // 32 threads
    if (t < 25) {
        float a = b2v[t];
        const float* x = tmpf + (size_t)nn * 128;
        for (int j = 0; j < 128; ++j) a += x[j] * W2[j * 25 + t];
        logits[nn * 25 + t] = a;
        sv[t] = (t >= 20) ? NEGV : a;
    }
    __syncthreads();
    if (last) {
        if (t == 0) {
            int bi = 0; float bv = sv[0];
            for (int k = 1; k < 25; ++k) if (sv[k] > bv) { bv = sv[k]; bi = k; }
            Sfin[nn] = mask[nn] ? bi : Sw[nn];
        }
    } else {
        if (t == 0) {
            float mx = sv[0];
            for (int k = 1; k < 25; ++k) mx = fmaxf(mx, sv[k]);
            float s = 0.f;
            for (int k = 0; k < 25; ++k) s += __expf(sv[k] - mx);
            red[0] = mx; red[1] = fast_rcp(s);
        }
        __syncthreads();
        if (t < 25) smooth[nn * 25 + t] = __expf(sv[t] - red[0]) * red[1];
    }
}

__global__ void k_xupdate(const unsigned char* __restrict__ mask,
                          const float* __restrict__ Xb, const float* __restrict__ Xc,
                          float* __restrict__ Xa) {
    int nn = blockIdx.x, t = threadIdx.x;             // 64
    if (t < 42) {
        size_t o = (size_t)nn * 42 + t;
        Xa[o] = mask[nn] ? Xb[o] : Xc[o];
    }
}

__global__ void k_finalize(const float* __restrict__ logits,
                           const float* __restrict__ predX,
                           const float* __restrict__ Hout,
                           const int* __restrict__ Sfin,
                           float* __restrict__ out, int out_size) {
    int i = blockIdx.x * blockDim.x + threadIdx.x;
    if (i >= out_size) return;
    float v = 0.f;
    if (i < 19200)        v = logits[i];
    else if (i < 51456)   v = predX[i - 19200];
    else if (i < 149760)  v = Hout[i - 51456];
    else { int k = i - 149760; if (k < 768) v = (float)Sfin[k]; }
    out[i] = v;
}

// ---------------- host ----------------
extern "C" void kernel_launch(void* const* d_in, const int* in_sizes, int n_in,
                              void* d_out, int out_size, void* d_ws, size_t ws_size,
                              hipStream_t stream) {
    (void)in_sizes; (void)n_in; (void)ws_size;
    const float*         X0    = (const float*)d_in[0];
    const int*           S0    = (const int*)d_in[1];
    const unsigned char* mask  = (const unsigned char*)d_in[2];
    const int*           pid   = (const int*)d_in[3];
    const unsigned char* amask = (const unsigned char*)d_in[7];
    const float* p_res_embed = (const float*)d_in[8];
    const float* p_pos_embed = (const float*)d_in[9];
    const float* p_atom_emb  = (const float*)d_in[10];
    const float* p_mem_W1 = (const float*)d_in[11];
    const float* p_mem_b1 = (const float*)d_in[12];
    const float* p_mem_W2 = (const float*)d_in[13];
    const float* p_mem_b2 = (const float*)d_in[14];
    const float* p_res_W1 = (const float*)d_in[15];
    const float* p_res_b1 = (const float*)d_in[16];
    const float* p_res_W2 = (const float*)d_in[17];
    const float* p_res_b2 = (const float*)d_in[18];
    const float* p_in_W   = (const float*)d_in[19];
    const float* p_in_b   = (const float*)d_in[20];
    const float* p_out_W  = (const float*)d_in[21];
    const float* p_out_b  = (const float*)d_in[22];
    const float* p_attn_W = (const float*)d_in[23];
    const float* p_rad_W  = (const float*)d_in[24];
    const float* p_rad_b  = (const float*)d_in[25];
    const float* p_edge_W1 = (const float*)d_in[26];
    const float* p_edge_b1 = (const float*)d_in[27];
    const float* p_edge_W2 = (const float*)d_in[28];
    const float* p_edge_b2 = (const float*)d_in[29];
    const float* p_coord_W1 = (const float*)d_in[30];
    const float* p_coord_b1 = (const float*)d_in[31];
    const float* p_coord_W2 = (const float*)d_in[32];
    const float* p_node_W1  = (const float*)d_in[33];
    const float* p_node_b1  = (const float*)d_in[34];
    const float* p_node_W2  = (const float*)d_in[35];
    const float* p_node_b2  = (const float*)d_in[36];

    char* wsb = (char*)d_ws;
    size_t off = 0;
    auto alloc = [&](size_t bytes) -> char* {
        char* p = wsb + off;
        off += (bytes + 255) & ~(size_t)255;
        return p;
    };
    const size_t XB = (size_t)NNODES * 42 * sizeof(float);
    const size_t HB = (size_t)NNODES * 128 * sizeof(float);
    const size_t HHB = (size_t)NNODES * 128 * sizeof(_Float16);

    float* Xa = (float*)alloc(XB);
    float* Xb = (float*)alloc(XB);
    float* Xc = (float*)alloc(XB);
    int*   Sw   = (int*)alloc(NNODES * sizeof(int));
    int*   Sfin = (int*)alloc(NNODES * sizeof(int));
    float* aw   = (float*)alloc((size_t)NNODES * 14 * sizeof(float));
    float* attnB = (float*)alloc((size_t)3 * NNODES * 56 * sizeof(float));
    float* cen     = (float*)alloc(32 * sizeof(float));
    float* centers = (float*)alloc(32 * sizeof(float));
    float* H0 = (float*)alloc(HB);         _Float16* H0h = (_Float16*)alloc(HHB);
    float* g_h = (float*)alloc(HB);        _Float16* g_h16 = (_Float16*)alloc(HHB);
    float* g_agg = (float*)alloc(HB);
    float* Hout = (float*)alloc(HB);       _Float16* HoutS16 = (_Float16*)alloc(HHB);
    _Float16* t1_16 = (_Float16*)alloc(HHB);
    float* tmpf = (float*)alloc(HB);
    float* smooth = (float*)alloc((size_t)NNODES * 25 * sizeof(float));
    float* logits = (float*)alloc((size_t)NNODES * 25 * sizeof(float));

    _Float16* swzMem1 = (_Float16*)alloc(16384 * 2);
    _Float16* swzMem2 = (_Float16*)alloc(16384 * 2);
    _Float16* swzRes1 = (_Float16*)alloc(16384 * 2);
    _Float16* swzInW  = (_Float16*)alloc(16384 * 2);
    _Float16* swzOutW = (_Float16*)alloc(16384 * 2);
    _Float16* swzE1 = (_Float16*)alloc((size_t)3 * 49152 * 2);
    _Float16* swzE2 = (_Float16*)alloc((size_t)3 * 16384 * 2);
    _Float16* swzC1 = (_Float16*)alloc((size_t)3 * 16384 * 2);
    _Float16* swzN1 = (_Float16*)alloc((size_t)3 * 32768 * 2);
    _Float16* swzN2 = (_Float16*)alloc((size_t)3 * 16384 * 2);

    auto swz = [&](const float* W, _Float16* dst, int K, int N) {
        int tot = K * N;
        k_swz<<<(tot + 255) / 256, 256, 0, stream>>>(W, dst, K, N);
    };
    swz(p_mem_W1, swzMem1, 128, 128);
    swz(p_mem_W2, swzMem2, 128, 128);
    swz(p_res_W1, swzRes1, 128, 128);
    swz(p_in_W,  swzInW,  128, 128);
    swz(p_out_W, swzOutW, 128, 128);
    for (int l = 0; l < 3; ++l) {
        swz(p_edge_W1  + (size_t)l * 49152, swzE1 + (size_t)l * 49152, 384, 128);
        swz(p_edge_W2  + (size_t)l * 16384, swzE2 + (size_t)l * 16384, 128, 128);
        swz(p_coord_W1 + (size_t)l * 16384, swzC1 + (size_t)l * 16384, 128, 128);
        swz(p_node_W1  + (size_t)l * 32768, swzN1 + (size_t)l * 32768, 256, 128);
        swz(p_node_W2  + (size_t)l * 16384, swzN2 + (size_t)l * 16384, 128, 128);
    }

    // ----- setup -----
    k_centroid<<<8, 96, 0, stream>>>(X0, mask, cen);
    k_initmask<<<NNODES, 64, 0, stream>>>(X0, S0, mask, amask, cen, Xa, Sw, aw);
    k_normstats<<<8, 96, 0, stream>>>(Xa, mask, centers);
    k_normalize<<<NNODES, 64, 0, stream>>>(Xa, centers);
    k_attn<<<NNODES, 64, 0, stream>>>(Sw, p_atom_emb, p_attn_W, aw, attnB);

    // ----- rounds -----
    for (int t = 0; t < 3; ++t) {
        (void)hipMemcpyAsync(Xc, Xa, XB, hipMemcpyDeviceToDevice, stream);
        k_embed<<<NNODES, 128, 0, stream>>>(Sw, mask, pid, p_res_embed, p_pos_embed,
                                            smooth, t > 0 ? 1 : 0, H0, H0h);
        if (t > 0) {
            k_lin128<<<48, 256, 0, stream>>>(HoutS16, swzMem1, p_mem_b1,
                                             nullptr, nullptr, t1_16, 1, 0);
            k_lin128<<<48, 256, 0, stream>>>(t1_16, swzMem2, p_mem_b2,
                                             H0, H0, H0h, 0, 0);
        }
        k_lin128<<<48, 256, 0, stream>>>(H0h, swzInW, p_in_b,
                                         nullptr, g_h, g_h16, 0, 0);
        float* Xi = Xa;
        float* Xo = Xb;
        for (int l = 0; l < 3; ++l) {
            k_edge_layer<<<NNODES, 256, 0, stream>>>(
                Xi, Xo, g_h16, attnB + (size_t)l * NNODES * 56, aw,
                p_rad_W + (size_t)l * 2048, p_rad_b + (size_t)l * 128,
                swzE1 + (size_t)l * 49152, p_edge_b1 + (size_t)l * 128,
                swzE2 + (size_t)l * 16384, p_edge_b2 + (size_t)l * 128,
                swzC1 + (size_t)l * 16384, p_coord_b1 + (size_t)l * 128,
                p_coord_W2 + (size_t)l * 128, g_agg);
            k_node_update<<<48, 256, 0, stream>>>(
                g_agg, g_h, g_h16,
                swzN1 + (size_t)l * 32768, p_node_b1 + (size_t)l * 128,
                swzN2 + (size_t)l * 16384, p_node_b2 + (size_t)l * 128);
            float* tmp = Xi; Xi = Xo; Xo = tmp;
        }
        // final pred X of this round is in Xb (layers: a->b, b->a, a->b)
        k_lin128<<<48, 256, 0, stream>>>(g_h16, swzOutW, p_out_b,
                                         nullptr, Hout, HoutS16, 0, 1);
        k_lin128<<<48, 256, 0, stream>>>(HoutS16, swzRes1, p_res_b1,
                                         nullptr, tmpf, nullptr, 1, 0);
        k_logits<<<NNODES, 32, 0, stream>>>(tmpf, p_res_W2, p_res_b2, mask, Sw,
                                            logits, smooth, Sfin, t == 2 ? 1 : 0);
        k_xupdate<<<NNODES, 64, 0, stream>>>(mask, Xb, Xc, Xa);
    }

    int tot = out_size;
    k_finalize<<<(tot + 255) / 256, 256, 0, stream>>>(logits, Xb, Hout, Sfin,
                                                      (float*)d_out, tot);
}
